// SAttention_90314572301028
// MI455X (gfx1250) — matrix operand order repaired and checked by reading, not compiled, not run on hardware
//
#include <hip/hip_runtime.h>
#include <hip/hip_bf16.h>
#include <math.h>

// ---------------------------------------------------------------------------
// Transformer block for MI455X (gfx1250, wave32, WMMA).
// bf16 WMMA 16x16x32 (fp32 accum) for all GEMMs; fp32 residual/LN path.
// GEMM: 256-thread block, tile 256(M)x64(N), B tile double-buffered in LDS
// via async global->LDS copies (ASYNCcnt) when available.
// Attention: flash-style over the stock axis, 32 query rows per wave.
// Workspace peak ~259 MB (offsets in kernel_launch).
// ---------------------------------------------------------------------------

#define D_MODEL 512
#define NSTK    512
#define TSEQ    64
#define NHEAD   8
#define DHEAD   64
#define NT      (NSTK * TSEQ)   // 32768 rows

typedef __bf16 bf16;
typedef __attribute__((ext_vector_type(16))) __bf16 v16bf;
typedef __attribute__((ext_vector_type(8)))  __bf16 v8bf;
typedef __attribute__((ext_vector_type(8)))  float  v8f;
typedef __attribute__((ext_vector_type(4)))  int    v4i;

union BF16x16 { v16bf v; v8bf h[2]; };

#if defined(__has_builtin)
#if __has_builtin(__builtin_amdgcn_global_load_async_to_lds_b128)
#define HAVE_ASYNC_LDS 1
#endif
#endif

// Stage one 16-byte chunk global -> LDS (async when the gfx1250 builtin exists).
// Builtin param 1 (from compiler diagnostic): non-const v4i* in AS(1).
static __device__ __forceinline__ void stage_b128(const bf16* g, bf16* l) {
#ifdef HAVE_ASYNC_LDS
    __builtin_amdgcn_global_load_async_to_lds_b128(
        (__attribute__((address_space(1))) v4i*)g,
        (__attribute__((address_space(3))) v4i*)l, /*offset=*/0, /*cpol=*/0);
#else
    uint4 t = *(const uint4*)g;
    *(uint4*)l = t;
#endif
}
static __device__ __forceinline__ void stage_wait() {
#ifdef HAVE_ASYNC_LDS
#if __has_builtin(__builtin_amdgcn_s_wait_asynccnt)
    __builtin_amdgcn_s_wait_asynccnt(0);
#else
    asm volatile("s_wait_asynccnt 0x0" ::: "memory");
#endif
#endif
}

// A-matrix fragment, 16(M) x 32(K) bf16, source row-major, ld elems.
// Lanes 0-15: rows, K={0..7,16..23}; lanes 16-31: same rows, K={8..15,24..31}.
static __device__ __forceinline__ v16bf load_a16x32(const bf16* p, int ld, int lane) {
    const int r  = lane & 15;
    const int hf = (lane >> 4) & 1;
    BF16x16 u;
    u.h[0] = *(const v8bf*)(p + (size_t)r * ld + hf * 8);
    u.h[1] = *(const v8bf*)(p + (size_t)r * ld + 16 + hf * 8);
    return u.v;
}

// B-matrix fragment, 32(K) x 16(N) bf16, from "Bt" layout: Bt[c][k] contiguous k.
// Lanes 0-15: col c, K={0..7,16..23}; lanes 16-31: col c, K={8..15,24..31}.
static __device__ __forceinline__ v16bf load_b32x16(const bf16* p, int ld, int lane) {
    const int c  = lane & 15;
    const int hf = (lane >> 4) & 1;
    BF16x16 u;
    u.h[0] = *(const v8bf*)(p + (size_t)c * ld + hf * 8);
    u.h[1] = *(const v8bf*)(p + (size_t)c * ld + 16 + hf * 8);
    return u.v;
}

static __device__ __forceinline__ v8f wmma_bf16(v16bf a, v16bf b, v8f c) {
    return __builtin_amdgcn_wmma_f32_16x16x32_bf16(
        false, a, false, b, (short)0, c, false, false);
}

static __device__ __forceinline__ float red_sum32(float v) {
#pragma unroll
    for (int m = 1; m < 32; m <<= 1) v += __shfl_xor(v, m, 32);
    return v;
}
static __device__ __forceinline__ float red_sum16(float v) {
#pragma unroll
    for (int m = 1; m < 16; m <<= 1) v += __shfl_xor(v, m, 32);
    return v;
}
static __device__ __forceinline__ float red_max16(float v) {
#pragma unroll
    for (int m = 1; m < 16; m <<= 1) v = fmaxf(v, __shfl_xor(v, m, 32));
    return v;
}

// ---------------------------------------------------------------------------
// Weight convert + transpose: wt[c*512 + k] = (bf16) w[k*512 + c]
// ---------------------------------------------------------------------------
__global__ __launch_bounds__(256) void convert_wt_kernel(const float* __restrict__ w,
                                                         bf16* __restrict__ wt) {
    const int idx = blockIdx.x * 256 + threadIdx.x;   // over 512*512
    const int c = idx >> 9;
    const int k = idx & 511;
    wt[idx] = (bf16)w[(size_t)k * D_MODEL + c];
}

// ---------------------------------------------------------------------------
// LayerNorm (pre): xn = LN(x; w,b) -> fp32 + bf16. One wave per 512-elem row.
// ---------------------------------------------------------------------------
__global__ __launch_bounds__(256) void ln_pre_kernel(const float* __restrict__ x,
                                                     const float* __restrict__ w,
                                                     const float* __restrict__ b,
                                                     float* __restrict__ yf,
                                                     bf16* __restrict__ yb) {
    const int row  = blockIdx.x * 8 + (threadIdx.x >> 5);
    const int lane = threadIdx.x & 31;
    const float4* xr = (const float4*)(x + (size_t)row * D_MODEL);
    float4 v[4];
    float s = 0.f, s2 = 0.f;
#pragma unroll
    for (int i = 0; i < 4; ++i) {
        v[i] = xr[lane + 32 * i];
        s  += v[i].x + v[i].y + v[i].z + v[i].w;
        s2 += v[i].x * v[i].x + v[i].y * v[i].y + v[i].z * v[i].z + v[i].w * v[i].w;
    }
    s = red_sum32(s); s2 = red_sum32(s2);
    const float mu   = s * (1.f / D_MODEL);
    const float rstd = rsqrtf(s2 * (1.f / D_MODEL) - mu * mu + 1e-5f);
#pragma unroll
    for (int i = 0; i < 4; ++i) {
        const int j = (lane + 32 * i) * 4;
        float4 o;
        o.x = (v[i].x - mu) * rstd * w[j + 0] + b[j + 0];
        o.y = (v[i].y - mu) * rstd * w[j + 1] + b[j + 1];
        o.z = (v[i].z - mu) * rstd * w[j + 2] + b[j + 2];
        o.w = (v[i].w - mu) * rstd * w[j + 3] + b[j + 3];
        ((float4*)(yf + (size_t)row * D_MODEL))[lane + 32 * i] = o;
        bf16* yo = yb + (size_t)row * D_MODEL + j;
        yo[0] = (bf16)o.x; yo[1] = (bf16)o.y; yo[2] = (bf16)o.z; yo[3] = (bf16)o.w;
    }
}

// ---------------------------------------------------------------------------
// Residual + double LN:  xt = xn + LN(att; w12,b12);  y = LN(xt; w2,b2)
// ---------------------------------------------------------------------------
__global__ __launch_bounds__(256) void ln_mid_kernel(const float* __restrict__ att,
                                                     const float* __restrict__ xn,
                                                     const float* __restrict__ w12,
                                                     const float* __restrict__ b12,
                                                     const float* __restrict__ w2,
                                                     const float* __restrict__ b2,
                                                     float* __restrict__ yf,
                                                     bf16* __restrict__ yb) {
    const int row  = blockIdx.x * 8 + (threadIdx.x >> 5);
    const int lane = threadIdx.x & 31;
    const float4* ar = (const float4*)(att + (size_t)row * D_MODEL);
    const float4* xr = (const float4*)(xn  + (size_t)row * D_MODEL);
    float4 a[4], xv[4];
    float s = 0.f, s2 = 0.f;
#pragma unroll
    for (int i = 0; i < 4; ++i) {
        a[i]  = ar[lane + 32 * i];
        xv[i] = xr[lane + 32 * i];
        s  += a[i].x + a[i].y + a[i].z + a[i].w;
        s2 += a[i].x * a[i].x + a[i].y * a[i].y + a[i].z * a[i].z + a[i].w * a[i].w;
    }
    s = red_sum32(s); s2 = red_sum32(s2);
    float mu   = s * (1.f / D_MODEL);
    float rstd = rsqrtf(s2 * (1.f / D_MODEL) - mu * mu + 1e-5f);
    float t = 0.f, t2 = 0.f;
#pragma unroll
    for (int i = 0; i < 4; ++i) {
        const int j = (lane + 32 * i) * 4;
        a[i].x = xv[i].x + ((a[i].x - mu) * rstd * w12[j + 0] + b12[j + 0]);
        a[i].y = xv[i].y + ((a[i].y - mu) * rstd * w12[j + 1] + b12[j + 1]);
        a[i].z = xv[i].z + ((a[i].z - mu) * rstd * w12[j + 2] + b12[j + 2]);
        a[i].w = xv[i].w + ((a[i].w - mu) * rstd * w12[j + 3] + b12[j + 3]);
        t  += a[i].x + a[i].y + a[i].z + a[i].w;
        t2 += a[i].x * a[i].x + a[i].y * a[i].y + a[i].z * a[i].z + a[i].w * a[i].w;
    }
    t = red_sum32(t); t2 = red_sum32(t2);
    mu   = t * (1.f / D_MODEL);
    rstd = rsqrtf(t2 * (1.f / D_MODEL) - mu * mu + 1e-5f);
#pragma unroll
    for (int i = 0; i < 4; ++i) {
        const int j = (lane + 32 * i) * 4;
        float4 o;
        o.x = (a[i].x - mu) * rstd * w2[j + 0] + b2[j + 0];
        o.y = (a[i].y - mu) * rstd * w2[j + 1] + b2[j + 1];
        o.z = (a[i].z - mu) * rstd * w2[j + 2] + b2[j + 2];
        o.w = (a[i].w - mu) * rstd * w2[j + 3] + b2[j + 3];
        ((float4*)(yf + (size_t)row * D_MODEL))[lane + 32 * i] = o;
        bf16* yo = yb + (size_t)row * D_MODEL + j;
        yo[0] = (bf16)o.x; yo[1] = (bf16)o.y; yo[2] = (bf16)o.z; yo[3] = (bf16)o.w;
    }
}

// ---------------------------------------------------------------------------
// Final: out = xt + LN(ffn; w22,b22)
// ---------------------------------------------------------------------------
__global__ __launch_bounds__(256) void ln_post_kernel(const float* __restrict__ ffn,
                                                      const float* __restrict__ xt,
                                                      const float* __restrict__ w22,
                                                      const float* __restrict__ b22,
                                                      float* __restrict__ out) {
    const int row  = blockIdx.x * 8 + (threadIdx.x >> 5);
    const int lane = threadIdx.x & 31;
    const float4* fr = (const float4*)(ffn + (size_t)row * D_MODEL);
    const float4* xr = (const float4*)(xt  + (size_t)row * D_MODEL);
    float4 f[4];
    float s = 0.f, s2 = 0.f;
#pragma unroll
    for (int i = 0; i < 4; ++i) {
        f[i] = fr[lane + 32 * i];
        s  += f[i].x + f[i].y + f[i].z + f[i].w;
        s2 += f[i].x * f[i].x + f[i].y * f[i].y + f[i].z * f[i].z + f[i].w * f[i].w;
    }
    s = red_sum32(s); s2 = red_sum32(s2);
    const float mu   = s * (1.f / D_MODEL);
    const float rstd = rsqrtf(s2 * (1.f / D_MODEL) - mu * mu + 1e-5f);
#pragma unroll
    for (int i = 0; i < 4; ++i) {
        const int j = (lane + 32 * i) * 4;
        float4 xv = xr[lane + 32 * i];
        float4 o;
        o.x = xv.x + ((f[i].x - mu) * rstd * w22[j + 0] + b22[j + 0]);
        o.y = xv.y + ((f[i].y - mu) * rstd * w22[j + 1] + b22[j + 1]);
        o.z = xv.z + ((f[i].z - mu) * rstd * w22[j + 2] + b22[j + 2]);
        o.w = xv.w + ((f[i].w - mu) * rstd * w22[j + 3] + b22[j + 3]);
        ((float4*)(out + (size_t)row * D_MODEL))[lane + 32 * i] = o;
    }
}

// ---------------------------------------------------------------------------
// bf16 WMMA GEMM: C[32768, 512] = A[32768,512] * Bt^T   (Bt is [512 c][512 k])
// Block 256 threads (8 waves); block tile 256(M) x 64(N); wave tile 32 x 64.
// B k-slice (64 cols x 32 k = 4 KB) double-buffered in LDS via async copies;
// padded row stride (40 elems = 80 B) -> conflict-free ds_load_b128.
// MODE epilogues: 1 bf16+bias+ReLU | 2 Q/K scatter | 3 V^T scatter | 4 fp32+bias
// ---------------------------------------------------------------------------
#define BPAD 40   // padded k-stride in LDS (elems)

template <int MODE>
__global__ __launch_bounds__(256) void gemm_ws_kernel(const bf16* __restrict__ A,
                                                      const bf16* __restrict__ Bt,
                                                      const float* __restrict__ bias,
                                                      void* __restrict__ outp) {
    __shared__ __align__(16) bf16 Bs[2][64 * BPAD];

    const int tid  = threadIdx.x;
    const int lane = tid & 31;
    const int wave = tid >> 5;
    const int m0 = blockIdx.y * 256 + wave * 32;
    const int n0 = blockIdx.x * 64;

    // staging: thread -> one 16B chunk of the 64x32 B slice
    const int sc = tid >> 2;            // 0..63 : column within tile
    const int sk = (tid & 3) * 8;       // 0,8,16,24 : k offset (elems)
    const bf16* gB = Bt + (size_t)(n0 + sc) * D_MODEL + sk;
    bf16* lB0 = &Bs[0][sc * BPAD + sk];
    bf16* lB1 = &Bs[1][sc * BPAD + sk];

    stage_b128(gB, lB0);                // k0 = 0 slice

    v8f acc[2][4] = {};
    const bf16* a_row = A + (size_t)m0 * D_MODEL;

    for (int k0 = 0, buf = 0; k0 < D_MODEL; k0 += 32, buf ^= 1) {
        stage_wait();
        __syncthreads();
        if (k0 + 32 < D_MODEL)
            stage_b128(gB + k0 + 32, buf ? lB0 : lB1);
        __builtin_prefetch(a_row + k0 + 32, 0, 1);

        const v16bf a0 = load_a16x32(a_row + k0, D_MODEL, lane);
        const v16bf a1 = load_a16x32(a_row + 16 * D_MODEL + k0, D_MODEL, lane);
        const bf16* bs = &Bs[buf][0];
#pragma unroll
        for (int cc = 0; cc < 4; ++cc) {
            const v16bf b = load_b32x16(bs + cc * 16 * BPAD, BPAD, lane);
            acc[0][cc] = wmma_bf16(a0, b, acc[0][cc]);
            acc[1][cc] = wmma_bf16(a1, b, acc[1][cc]);
        }
    }

    const int half = lane >> 4, c_in = lane & 15;
#pragma unroll
    for (int g = 0; g < 2; ++g) {
#pragma unroll
        for (int cc = 0; cc < 4; ++cc) {
#pragma unroll
            for (int i = 0; i < 8; ++i) {
                const int m = m0 + g * 16 + i + 8 * half;
                const int n = n0 + cc * 16 + c_in;
                float val = acc[g][cc][i];
                if (MODE == 1) {
                    val = fmaxf(val + bias[n], 0.f);
                    ((bf16*)outp)[(size_t)m * D_MODEL + n] = (bf16)val;
                } else if (MODE == 2) {
                    const int ns = m >> 6, t = m & 63, hh = n >> 6, di = n & 63;
                    ((bf16*)outp)[((size_t)(t * NHEAD + hh) * NSTK + ns) * DHEAD + di] = (bf16)val;
                } else if (MODE == 3) {
                    const int ns = m >> 6, t = m & 63, hh = n >> 6, di = n & 63;
                    ((bf16*)outp)[((size_t)(t * NHEAD + hh) * DHEAD + di) * NSTK + ns] = (bf16)val;
                } else if (MODE == 4) {
                    ((float*)outp)[(size_t)m * D_MODEL + n] = val + bias[n];
                } else {
                    ((float*)outp)[(size_t)m * D_MODEL + n] = val;
                }
            }
        }
    }
}

// ---------------------------------------------------------------------------
// Flash-style attention over the stock axis, per (t, h).
// Grid: (T*H, N/256); block 256 = 8 waves; each wave = 32 query rows (2 groups).
// q,k: [T*H][N][dh] bf16; vt: [T*H][dh][N] bf16; out att_o: [N][T][d] fp32.
// ---------------------------------------------------------------------------
__global__ __launch_bounds__(256) void attn_kernel(const bf16* __restrict__ q,
                                                   const bf16* __restrict__ k,
                                                   const bf16* __restrict__ vt,
                                                   float* __restrict__ att_o) {
    __shared__ __align__(16) bf16 plds[8][2][16 * BPAD];   // per-wave prob tiles

    const int lane = threadIdx.x & 31;
    const int wave = threadIdx.x >> 5;
    const int th = blockIdx.x;                 // t*NHEAD + h
    const int t = th >> 3, h = th & 7;
    const int m0 = blockIdx.y * 256 + wave * 32;

    const bf16* Qp = q  + (size_t)th * NSTK * DHEAD;
    const bf16* Kp = k  + (size_t)th * NSTK * DHEAD;
    const bf16* Vp = vt + (size_t)th * DHEAD * NSTK;

    v16bf aq[2][2];
#pragma unroll
    for (int g = 0; g < 2; ++g) {
        aq[g][0] = load_a16x32(Qp + (size_t)(m0 + g * 16) * DHEAD,      DHEAD, lane);
        aq[g][1] = load_a16x32(Qp + (size_t)(m0 + g * 16) * DHEAD + 32, DHEAD, lane);
    }

    v8f O[2][4] = {};
    float rowM[2][8], rowL[2][8];
#pragma unroll
    for (int g = 0; g < 2; ++g)
#pragma unroll
        for (int i = 0; i < 8; ++i) { rowM[g][i] = -1e30f; rowL[g][i] = 0.f; }

    const float invT = 0.125f;                 // 1 / sqrt(d/H) = 1/8
    const int half = lane >> 4, ccol = lane & 15;

    for (int j0 = 0; j0 < NSTK; j0 += 32) {
        // ---- score tiles: s[g][ct] (rows m0+g*16.., key cols j0+ct*16..)
        v8f s[2][2];
        {
            v16bf b0 = load_b32x16(Kp + (size_t)j0 * DHEAD,      DHEAD, lane);
            v16bf b1 = load_b32x16(Kp + (size_t)j0 * DHEAD + 32, DHEAD, lane);
            v8f z = {};
            s[0][0] = wmma_bf16(aq[0][1], b1, wmma_bf16(aq[0][0], b0, z));
            s[1][0] = wmma_bf16(aq[1][1], b1, wmma_bf16(aq[1][0], b0, z));
            b0 = load_b32x16(Kp + (size_t)(j0 + 16) * DHEAD,      DHEAD, lane);
            b1 = load_b32x16(Kp + (size_t)(j0 + 16) * DHEAD + 32, DHEAD, lane);
            s[0][1] = wmma_bf16(aq[0][1], b1, wmma_bf16(aq[0][0], b0, z));
            s[1][1] = wmma_bf16(aq[1][1], b1, wmma_bf16(aq[1][0], b0, z));
        }
        // ---- online softmax per row group; transpose probs through LDS
#pragma unroll
        for (int g = 0; g < 2; ++g) {
            float p0[8], p1[8], f[8];
#pragma unroll
            for (int i = 0; i < 8; ++i) {
                const float a  = s[g][0][i] * invT;
                const float bb = s[g][1][i] * invT;
                const float mx = red_max16(fmaxf(a, bb));
                const float nm = fmaxf(rowM[g][i], mx);
                f[i]  = __expf(rowM[g][i] - nm);
                p0[i] = __expf(a - nm);
                p1[i] = __expf(bb - nm);
                const float rs = red_sum16(p0[i] + p1[i]);
                rowL[g][i] = rowL[g][i] * f[i] + rs;
                rowM[g][i] = nm;
            }
#pragma unroll
            for (int cc = 0; cc < 4; ++cc)
#pragma unroll
                for (int i = 0; i < 8; ++i) O[g][cc][i] *= f[i];
            bf16* pw = &plds[wave][g][0];
#pragma unroll
            for (int i = 0; i < 8; ++i) {
                pw[(i + 8 * half) * BPAD + ccol]      = (bf16)p0[i];
                pw[(i + 8 * half) * BPAD + 16 + ccol] = (bf16)p1[i];
            }
        }
        const v16bf pa0 = load_a16x32(&plds[wave][0][0], BPAD, lane);
        const v16bf pa1 = load_a16x32(&plds[wave][1][0], BPAD, lane);

        // ---- O += P * V (B from transposed V: contiguous per lane)
#pragma unroll
        for (int cc = 0; cc < 4; ++cc) {
            const v16bf bv = load_b32x16(Vp + (size_t)(cc * 16) * NSTK + j0, NSTK, lane);
            O[0][cc] = wmma_bf16(pa0, bv, O[0][cc]);
            O[1][cc] = wmma_bf16(pa1, bv, O[1][cc]);
        }
    }

    // ---- normalize and scatter to att_o [N][T][d]
#pragma unroll
    for (int g = 0; g < 2; ++g) {
        float inv[8];
#pragma unroll
        for (int i = 0; i < 8; ++i) inv[i] = 1.f / rowL[g][i];
#pragma unroll
        for (int cc = 0; cc < 4; ++cc) {
#pragma unroll
            for (int i = 0; i < 8; ++i) {
                const int n   = m0 + g * 16 + i + 8 * half;
                const int col = h * DHEAD + cc * 16 + ccol;
                att_o[((size_t)n * TSEQ + t) * D_MODEL + col] = O[g][cc][i] * inv[i];
            }
        }
    }
}

// ---------------------------------------------------------------------------
// Host-side orchestration
// ---------------------------------------------------------------------------
extern "C" void kernel_launch(void* const* d_in, const int* in_sizes, int n_in,
                              void* d_out, int out_size, void* d_ws, size_t ws_size,
                              hipStream_t stream) {
    (void)in_sizes; (void)n_in; (void)out_size; (void)ws_size;

    const float* x    = (const float*)d_in[0];
    const float* qw   = (const float*)d_in[1];
    const float* kw   = (const float*)d_in[2];
    const float* vw   = (const float*)d_in[3];
    const float* n1w  = (const float*)d_in[4];
    const float* n1b  = (const float*)d_in[5];
    const float* n12w = (const float*)d_in[6];
    const float* n12b = (const float*)d_in[7];
    const float* n2w  = (const float*)d_in[8];
    const float* n2b  = (const float*)d_in[9];
    const float* n22w = (const float*)d_in[10];
    const float* n22b = (const float*)d_in[11];
    const float* w1   = (const float*)d_in[12];
    const float* b1   = (const float*)d_in[13];
    const float* w2   = (const float*)d_in[14];
    const float* b2   = (const float*)d_in[15];

    char* ws = (char*)d_ws;
    const size_t MB = 1ull << 20;
    // Phase 1 buffers
    float* xn_f  = (float*)(ws + 0 * MB);      // 64 MB  LN1 out (fp32)
    bf16*  xn_b  = (bf16*) (ws + 64 * MB);     // 32 MB  LN1 out (bf16)
    bf16*  q_b   = (bf16*) (ws + 96 * MB);     // 32 MB  [T*H][N][dh]
    bf16*  k_b   = (bf16*) (ws + 128 * MB);    // 32 MB
    bf16*  v_b   = (bf16*) (ws + 160 * MB);    // 32 MB  [T*H][dh][N]
    float* att_o = (float*)(ws + 192 * MB);    // 64 MB  attention out [N][T][d]
    bf16*  wq_t  = (bf16*) (ws + 256 * MB);    // 0.5 MB each, transposed bf16
    bf16*  wk_t  = (bf16*) (ws + 256 * MB + 512 * 1024);
    bf16*  wv_t  = (bf16*) (ws + 257 * MB);
    bf16*  w1_t  = (bf16*) (ws + 257 * MB + 512 * 1024);
    bf16*  w2_t  = (bf16*) (ws + 258 * MB);
    // Phase 2 reuse
    float* xtn_f  = (float*)(ws + 96 * MB);    // 64 MB  (over q_b + k_b)
    bf16*  xtn_b  = (bf16*) (ws + 160 * MB);   // 32 MB  (over v_b)
    bf16*  h1_b   = (bf16*) (ws + 0 * MB);     // 32 MB  (over xn_f)
    float* ffn2_f = (float*)(ws + 192 * MB);   // 64 MB  (over att_o)

    const dim3 blk(256);
    const dim3 gW(512 * 512 / 256);            // weight convert
    const dim3 gL(NT / 8);                     // layernorm rows
    const dim3 gG(D_MODEL / 64, NT / 256);     // GEMM tiles
    const dim3 gA(TSEQ * NHEAD, NSTK / 256);   // attention

    convert_wt_kernel<<<gW, blk, 0, stream>>>(qw, wq_t);
    convert_wt_kernel<<<gW, blk, 0, stream>>>(kw, wk_t);
    convert_wt_kernel<<<gW, blk, 0, stream>>>(vw, wv_t);
    convert_wt_kernel<<<gW, blk, 0, stream>>>(w1, w1_t);
    convert_wt_kernel<<<gW, blk, 0, stream>>>(w2, w2_t);

    ln_pre_kernel<<<gL, blk, 0, stream>>>(x, n1w, n1b, xn_f, xn_b);

    gemm_ws_kernel<2><<<gG, blk, 0, stream>>>(xn_b, wq_t, nullptr, q_b);
    gemm_ws_kernel<2><<<gG, blk, 0, stream>>>(xn_b, wk_t, nullptr, k_b);
    gemm_ws_kernel<3><<<gG, blk, 0, stream>>>(xn_b, wv_t, nullptr, v_b);

    attn_kernel<<<gA, blk, 0, stream>>>(q_b, k_b, v_b, att_o);

    ln_mid_kernel<<<gL, blk, 0, stream>>>(att_o, xn_f, n12w, n12b, n2w, n2b,
                                          xtn_f, xtn_b);

    gemm_ws_kernel<1><<<gG, blk, 0, stream>>>(xtn_b, w1_t, b1, h1_b);
    gemm_ws_kernel<4><<<gG, blk, 0, stream>>>(h1_b, w2_t, b2, ffn2_f);

    ln_post_kernel<<<gL, blk, 0, stream>>>(ffn2_f, xtn_f, n22w, n22b, (float*)d_out);
}
